// MeshUpConv_49383533969437
// MI455X (gfx1250) — matrix operand order repaired and checked by reading, not compiled, not run on hardware
//
#include <hip/hip_runtime.h>
#include <hip/hip_bf16.h>

// ---------------------------------------------------------------------------
// Problem constants
// ---------------------------------------------------------------------------
#define B_   8
#define E_   16384
#define O_   256
#define EPS_ 1e-5f
#define EM_  64          // edges (M) per block
#define ET_  (EM_ / 16)  // 16-edge WMMA tiles per block

typedef __attribute__((ext_vector_type(16))) __bf16 v16bf;
typedef __attribute__((ext_vector_type(8)))  float  v8f;

// ---------------------------------------------------------------------------
// bf16 helpers (RNE)
// ---------------------------------------------------------------------------
__device__ __forceinline__ unsigned short f2bf(float f) {
  unsigned u = __float_as_uint(f);
  u += 0x7fffu + ((u >> 16) & 1u);
  return (unsigned short)(u >> 16);
}
__device__ __forceinline__ float bf2f(unsigned short h) {
  return __uint_as_float(((unsigned)h) << 16);
}
__device__ __forceinline__ unsigned pk2(float a, float b) {
  return (unsigned)f2bf(a) | ((unsigned)f2bf(b) << 16);
}
__device__ __forceinline__ float ldin(const float* p)          { return *p; }
__device__ __forceinline__ float ldin(const unsigned short* p) { return bf2f(*p); }

// ---------------------------------------------------------------------------
// Weight repack: W[O][C][5] f32  ->  Wt[k][cc][ot][lane][i] packed bf16 pairs
// laid out as 16-bit B-matrix (32x16) fragments in contiguous-k order:
//   n = lane&15 ; K = 16*(lane>>4) + 2*i (+1 in hi half of dword)
// ---------------------------------------------------------------------------
__global__ void prep_w_kernel(const float* __restrict__ W,
                              unsigned* __restrict__ Wt, int C) {
  const int nCC   = C >> 5;
  const int total = 5 * nCC * 16 * 32 * 8;
  int idx = blockIdx.x * blockDim.x + threadIdx.x;
  if (idx >= total) return;
  const int i    = idx & 7;
  const int lane = (idx >> 3) & 31;
  const int ot   = (idx >> 8) & 15;
  const int cc   = (idx >> 12) % nCC;
  const int k    = (idx >> 12) / nCC;
  const int o    = ot * 16 + (lane & 15);
  const int K    = 16 * (lane >> 4) + 2 * i;
  const int c    = cc * 32 + K;
  const float w0 = W[((size_t)o * C + c) * 5 + k];
  const float w1 = W[((size_t)o * C + c + 1) * 5 + k];
  Wt[idx] = pk2(w0, w1);
}

// from_down f32 [B][256][E] -> xcat bf16 channels [256..511]
__global__ void cast_down_kernel(const float* __restrict__ fd,
                                 unsigned short* __restrict__ xcat) {
  size_t i = (size_t)blockIdx.x * blockDim.x + threadIdx.x;   // B*256*E
  const int b = (int)(i >> 22);                               // 256*E = 2^22
  const size_t rem = i & (((size_t)1 << 22) - 1);             // c*E + e
  xcat[((size_t)b * 512 + 256) * (size_t)E_ + rem] = f2bf(fd[i]);
}

// ---------------------------------------------------------------------------
// Mesh conv via WMMA.
// Block = 256 threads (8 wave32), one (batch, 64-edge tile). Each wave owns
// two 16-wide o-tiles across 4 e-tiles (8 accumulators); loop over
// 32-channel chunks x 5 features. B-fragments are reused across 4 e-tiles,
// cutting weight L2 traffic 4x vs a 16-edge M-tile.
// ---------------------------------------------------------------------------
template <typename TIn, bool OUT_BF16>
__global__ __launch_bounds__(256)
void mesh_conv_wmma(const TIn* __restrict__ x, const int* __restrict__ ei,
                    const unsigned* __restrict__ Wt,
                    const float* __restrict__ bias,
                    void* __restrict__ out, int C, int out_ch_stride) {
  const int eblk = blockIdx.x;        // 64-edge tile
  const int b    = blockIdx.y;
  const int nCC  = C >> 5;
  const int t    = threadIdx.x;
  const int wave = t >> 5;
  const int lane = t & 31;
  const int e0   = eblk * EM_;

  __shared__ __align__(32) unsigned short a_tile[5][ET_][32 * 16];
  __shared__ int nbr[EM_][4];

  // one neighbor index per thread (64 edges x 4)
  nbr[t >> 2][t & 3] = ei[(((size_t)b * E_) + e0 + (t >> 2)) * 4 + (t & 3)];
  __syncthreads();

  v8f acc[2][ET_] = {};

  for (int cc = 0; cc < nCC; ++cc) {
    // ---- build 5 feature A-tiles (64 edges x 32 channels, bf16) in LDS ----
#pragma unroll
    for (int s = 0; s < EM_ * 32 / 256; ++s) {
      const int p = t + 256 * s;          // 2048 (e,c) pairs
      const int e = p >> 5;               // 0..63
      const int c = p & 31;
      const TIn* base = x + ((size_t)b * C + (size_t)(cc * 32 + c)) * E_;
      const float self = ldin(base + e0 + e);
      const float n1 = ldin(base + nbr[e][0]);
      const float n2 = ldin(base + nbr[e][1]);
      const float n3 = ldin(base + nbr[e][2]);
      const float n4 = ldin(base + nbr[e][3]);
      // 16-bit A-matrix (16x32) in contiguous-k order (as Wt): lane,vgpr,halfword for K=c
      const int ln  = (e & 15) + ((c >> 4) << 4);
      const int idx = ln * 16 + (c & 15);
      const int et  = e >> 4;
      a_tile[0][et][idx] = f2bf(self);
      a_tile[1][et][idx] = f2bf(n1 + n3);
      a_tile[2][et][idx] = f2bf(n2 + n4);
      a_tile[3][et][idx] = f2bf(fabsf(n1 - n3));
      a_tile[4][et][idx] = f2bf(fabsf(n2 - n4));
    }
    __syncthreads();

    if (cc + 1 < nCC)   // warm GL2 for next chunk's B fragments
      __builtin_prefetch(Wt + (((size_t)cc + 1) * 16 * 32 * 8), 0, 1);

#pragma unroll
    for (int k = 0; k < 5; ++k) {
      const unsigned* wb =
          Wt + ((((size_t)k * nCC + cc) * 16) * 32 + lane) * 8;
      const v16bf bf0 = *reinterpret_cast<const v16bf*>(wb + (size_t)(wave * 2)     * 256);
      const v16bf bf1 = *reinterpret_cast<const v16bf*>(wb + (size_t)(wave * 2 + 1) * 256);
#pragma unroll
      for (int et = 0; et < ET_; ++et) {
        const v16bf af =
            *reinterpret_cast<const v16bf*>(&a_tile[k][et][lane * 16]);
        acc[0][et] = __builtin_amdgcn_wmma_f32_16x16x32_bf16(
            false, af, false, bf0, (short)0, acc[0][et], false, false);
        acc[1][et] = __builtin_amdgcn_wmma_f32_16x16x32_bf16(
            false, af, false, bf1, (short)0, acc[1][et], false, false);
      }
    }
    __syncthreads();
  }

  // ---- epilogue: C/D layout -> e = r + 8*(lane>=16), o = lane&15 ----
#pragma unroll
  for (int j = 0; j < 2; ++j) {
    const int o    = (wave * 2 + j) * 16 + (lane & 15);
    const float bv = bias[o];
    const size_t rowoff = ((size_t)b * out_ch_stride + o) * (size_t)E_;
#pragma unroll
    for (int et = 0; et < ET_; ++et) {
      const int eb = e0 + et * 16 + ((lane >> 4) << 3);
      if constexpr (OUT_BF16) {
        uint4 u;
        u.x = pk2(acc[j][et][0] + bv, acc[j][et][1] + bv);
        u.y = pk2(acc[j][et][2] + bv, acc[j][et][3] + bv);
        u.z = pk2(acc[j][et][4] + bv, acc[j][et][5] + bv);
        u.w = pk2(acc[j][et][6] + bv, acc[j][et][7] + bv);
        *reinterpret_cast<uint4*>((unsigned short*)out + rowoff + eb) = u;
      } else {
        float* op = (float*)out + rowoff + eb;
        *reinterpret_cast<float4*>(op) =
            make_float4(acc[j][et][0] + bv, acc[j][et][1] + bv,
                        acc[j][et][2] + bv, acc[j][et][3] + bv);
        *reinterpret_cast<float4*>(op + 4) =
            make_float4(acc[j][et][4] + bv, acc[j][et][5] + bv,
                        acc[j][et][6] + bv, acc[j][et][7] + bv);
      }
    }
  }
}

// ---------------------------------------------------------------------------
// Instance-norm stats: one block per (b,o) row of E elements
// ---------------------------------------------------------------------------
__global__ __launch_bounds__(256)
void row_stats_kernel(const float* __restrict__ t, float* __restrict__ mean,
                      float* __restrict__ inv) {
  const int row = blockIdx.x;
  const float* p = t + (size_t)row * E_;
  float s = 0.f, sq = 0.f;
  for (int i = threadIdx.x; i < E_; i += 256) {
    const float v = p[i];
    s += v; sq += v * v;
  }
  __shared__ float sh1[256], sh2[256];
  sh1[threadIdx.x] = s; sh2[threadIdx.x] = sq;
  __syncthreads();
  for (int o = 128; o > 0; o >>= 1) {
    if (threadIdx.x < o) {
      sh1[threadIdx.x] += sh1[threadIdx.x + o];
      sh2[threadIdx.x] += sh2[threadIdx.x + o];
    }
    __syncthreads();
  }
  if (threadIdx.x == 0) {
    const float m = sh1[0] * (1.f / E_);
    const float v = sh2[0] * (1.f / E_) - m * m;
    mean[row] = m;
    inv[row]  = rsqrtf(v + EPS_);
  }
}

__global__ void norm_relu_kernel(const float* __restrict__ t,
                                 const float* __restrict__ mean,
                                 const float* __restrict__ inv,
                                 unsigned short* __restrict__ out) {
  const size_t i = (size_t)blockIdx.x * blockDim.x + threadIdx.x;
  const int row = (int)(i >> 14);
  const float y = (t[i] - mean[row]) * inv[row];
  out[i] = f2bf(fmaxf(y, 0.f));
}

__global__ void norm_res_kernel(const float* __restrict__ t,
                                const float* __restrict__ mean,
                                const float* __restrict__ inv,
                                const unsigned short* __restrict__ prev,
                                unsigned short* __restrict__ out_bf,
                                float* __restrict__ out_f) {
  const size_t i = (size_t)blockIdx.x * blockDim.x + threadIdx.x;
  const int row = (int)(i >> 14);
  float y = (t[i] - mean[row]) * inv[row] + bf2f(prev[i]);
  y = fmaxf(y, 0.f);
  if (out_bf) out_bf[i] = f2bf(y);
  if (out_f)  out_f[i]  = y;
}

// ---------------------------------------------------------------------------
// Launcher
// ---------------------------------------------------------------------------
extern "C" void kernel_launch(void* const* d_in, const int* in_sizes, int n_in,
                              void* d_out, int out_size, void* d_ws, size_t ws_size,
                              hipStream_t stream) {
  (void)in_sizes; (void)n_in; (void)out_size; (void)ws_size;
  const float* from_up   = (const float*)d_in[0];
  const float* from_down = (const float*)d_in[1];
  const int*   edge_idx  = (const int*)d_in[2];
  const float* W_up = (const float*)d_in[3];  const float* b_up = (const float*)d_in[4];
  const float* W1   = (const float*)d_in[5];  const float* b1   = (const float*)d_in[6];
  const float* W2a  = (const float*)d_in[7];  const float* b2a  = (const float*)d_in[8];
  const float* W2b  = (const float*)d_in[9];  const float* b2b  = (const float*)d_in[10];
  float* out = (float*)d_out;

  char* ws = (char*)d_ws;
  size_t off = 0;
  auto alloc = [&](size_t bytes) -> void* {
    void* p = ws + off;
    off = (off + bytes + 255) & ~(size_t)255;
    return p;
  };
  const size_t actF32 = (size_t)B_ * O_ * E_ * sizeof(float);          // 134 MB
  const size_t actBF  = (size_t)B_ * O_ * E_ * sizeof(unsigned short); // 67 MB

  unsigned* WtU  = (unsigned*)alloc((size_t)5 * 4  * 4096 * 4);
  unsigned* Wt1  = (unsigned*)alloc((size_t)5 * 16 * 4096 * 4);
  unsigned* Wt2a = (unsigned*)alloc((size_t)5 * 8  * 4096 * 4);
  unsigned* Wt2b = (unsigned*)alloc((size_t)5 * 8  * 4096 * 4);
  unsigned short* xcat = (unsigned short*)alloc((size_t)B_ * 512 * E_ * 2);
  float*          tbuf = (float*)alloc(actF32);
  unsigned short* x1   = (unsigned short*)alloc(actBF);
  unsigned short* x2   = (unsigned short*)alloc(actBF);
  float* meanb = (float*)alloc((size_t)B_ * O_ * 4);
  float* invb  = (float*)alloc((size_t)B_ * O_ * 4);

  const dim3 blk(256);
  const dim3 convGrid(E_ / EM_, B_);
  const int  nElem     = B_ * O_ * E_;
  const int  elemGrid  = nElem / 256;
  const int  statsGrid = B_ * O_;

  // 1) repack weights to WMMA B-fragment layout (bf16)
  prep_w_kernel<<<(5 * 4  * 4096 + 255) / 256, blk, 0, stream>>>(W_up, WtU, 128);
  prep_w_kernel<<<(5 * 16 * 4096 + 255) / 256, blk, 0, stream>>>(W1,   Wt1, 512);
  prep_w_kernel<<<(5 * 8  * 4096 + 255) / 256, blk, 0, stream>>>(W2a,  Wt2a, 256);
  prep_w_kernel<<<(5 * 8  * 4096 + 255) / 256, blk, 0, stream>>>(W2b,  Wt2b, 256);

  // 2) xcat[256..511] = bf16(from_down)
  cast_down_kernel<<<elemGrid, blk, 0, stream>>>(from_down, xcat);

  // 3) conv1: f32 from_up (C=128) -> xcat[0..255] bf16 (channel stride 512)
  mesh_conv_wmma<float, true><<<convGrid, blk, 0, stream>>>(
      from_up, edge_idx, WtU, b_up, xcat, 128, 512);

  // 4) conv2: xcat bf16 (C=512) -> tbuf f32
  mesh_conv_wmma<unsigned short, false><<<convGrid, blk, 0, stream>>>(
      xcat, edge_idx, Wt1, b1, tbuf, 512, 256);

  // 5) x1 = relu(instnorm(tbuf)) (bf16)
  row_stats_kernel<<<statsGrid, blk, 0, stream>>>(tbuf, meanb, invb);
  norm_relu_kernel<<<elemGrid, blk, 0, stream>>>(tbuf, meanb, invb, x1);

  // 6) residual block a: x2 = relu(instnorm(conv(x1)) + x1)
  mesh_conv_wmma<unsigned short, false><<<convGrid, blk, 0, stream>>>(
      x1, edge_idx, Wt2a, b2a, tbuf, 256, 256);
  row_stats_kernel<<<statsGrid, blk, 0, stream>>>(tbuf, meanb, invb);
  norm_res_kernel<<<elemGrid, blk, 0, stream>>>(tbuf, meanb, invb, x1, x2, nullptr);

  // 7) residual block b: out = relu(instnorm(conv(x2)) + x2)  (f32 -> d_out)
  mesh_conv_wmma<unsigned short, false><<<convGrid, blk, 0, stream>>>(
      x2, edge_idx, Wt2b, b2b, tbuf, 256, 256);
  row_stats_kernel<<<statsGrid, blk, 0, stream>>>(tbuf, meanb, invb);
  norm_res_kernel<<<elemGrid, blk, 0, stream>>>(tbuf, meanb, invb, x2, nullptr, out);
}
